// CognitiveSpectralLM_24739011625056
// MI455X (gfx1250) — matrix operand order repaired and checked by reading, not compiled, not run on hardware
//
#include <hip/hip_runtime.h>
#include <hip/hip_bf16.h>
#include <math.h>

// ---------------- constants ----------------
#define NTOK   32768      // B*T
#define DMODEL 1024
#define HALF   512
#define NCODES 4096

typedef __attribute__((ext_vector_type(16))) _Float16 v16h;
typedef __attribute__((ext_vector_type(8)))  float    v8f;

union H16 { v16h v; _Float16 h[16]; uint4 q[2]; };

// ---------------- workspace layout (bytes) ----------------
#define OFF_WC     ((size_t)0)                       // 512*1024 f16  = 1 MB
#define OFF_WS     ((size_t)1048576)                 // 1 MB
#define OFF_WCOMB  ((size_t)2097152)                 // 1024*1024 f16 = 2 MB
#define OFF_CBC    ((size_t)4194304)                 // 4096*512 f16  = 4 MB
#define OFF_CBS    ((size_t)8388608)                 // 4 MB
#define OFF_CNT    ((size_t)12582912)                // 32768*512 f16 = 32 MB
#define OFF_STY    ((size_t)46137344)                // 32 MB
#define OFF_QUANT  ((size_t)79691776)                // 32768*1024 f16 = 64 MB
#define OFF_MU     ((size_t)146800640)               // 32768 f32
#define OFF_RS     ((size_t)146931712)
#define OFF_IDXC   ((size_t)147062784)               // 32768 i32
#define OFF_IDXS   ((size_t)147193856)
#define OFF_ACC    ((size_t)147324928)               // 4 f32 accumulators

// ---------------- d_out layout (float elements) ----------------
#define OUT_TOTAL    ((size_t)33554432)
#define OUT_IDXC     ((size_t)33554433)
#define OUT_IDXS     ((size_t)33587201)
#define OUT_DIS      ((size_t)33619969)

// ---------------- WMMA helper ----------------
__device__ __forceinline__ v8f wmma32(v16h a, v16h b, v8f c) {
  return __builtin_amdgcn_wmma_f32_16x16x32_f16(false, a, false, b, (short)0, c,
                                                false, false);
}

// A fragment: row-major f16 A[M,K] (lda halfs). lane m=l&15, h=l>>4:
// elems 0..7 = A[row0+m][k+8h .. +7], elems 8..15 = A[row0+m][k+16+8h .. +7]
__device__ __forceinline__ v16h load_a_f16(const _Float16* __restrict__ A,
                                           int lda, int row0, int k, int lane) {
  int m = lane & 15, hh = lane >> 4;
  const _Float16* p = A + (size_t)(row0 + m) * lda + k + 8 * hh;
  H16 u;
  u.q[0] = *(const uint4*)p;
  u.q[1] = *(const uint4*)(p + 16);
  return u.v;
}

// B fragment: computing A*W^T with row-major W[Nout,K] (ldw halfs).
// lane n=l&15, h=l>>4: elems 0..7 = W[col0+n][k+8h .. +7], elems 8..15 = W[col0+n][k+16+8h .. +7].
__device__ __forceinline__ v16h load_b_f16(const _Float16* __restrict__ W,
                                           int ldw, int col0, int k, int lane) {
  int n = lane & 15, hh = lane >> 4;
  const _Float16* p = W + (size_t)(col0 + n) * ldw + k + 8 * hh;
  H16 u;
  u.q[0] = *(const uint4*)p;
  u.q[1] = *(const uint4*)(p + 16);
  return u.v;
}

// A fragment with fused LayerNorm affine from fp32 x row; gamma/beta
// already loaded (shared across row-groups within a k-step).
__device__ __forceinline__ v16h make_a_ln(const float* __restrict__ xrow,
                                          float muv, float rsv,
                                          const float4* gv, const float4* bv,
                                          int k, int hh) {
  H16 u;
  int kk = k + 8 * hh;
#pragma unroll
  for (int part = 0; part < 2; ++part) {
    const float4* xp = (const float4*)(xrow + kk + part * 16);
#pragma unroll
    for (int q = 0; q < 2; ++q) {
      float4 xv = xp[q];
      float4 g = gv[part * 2 + q], b = bv[part * 2 + q];
      int e = part * 8 + q * 4;
      u.h[e + 0] = (_Float16)((xv.x - muv) * rsv * g.x + b.x);
      u.h[e + 1] = (_Float16)((xv.y - muv) * rsv * g.y + b.y);
      u.h[e + 2] = (_Float16)((xv.z - muv) * rsv * g.z + b.z);
      u.h[e + 3] = (_Float16)((xv.w - muv) * rsv * g.w + b.w);
    }
  }
  return u.v;
}

// ---------------- small kernels ----------------
__global__ __launch_bounds__(256) void k_convert_f16(const float* __restrict__ in,
                                                     _Float16* __restrict__ out,
                                                     int n) {
  int i = blockIdx.x * 256 + threadIdx.x;
  if (i < n) out[i] = (_Float16)in[i];
}

// L2-normalize codebook rows -> f16. One wave per row.
__global__ __launch_bounds__(256) void k_norm_cb(const float* __restrict__ cb,
                                                 _Float16* __restrict__ out) {
  int lane = threadIdx.x & 31;
  int row = blockIdx.x * 8 + (threadIdx.x >> 5);
  const float* p = cb + (size_t)row * HALF;
  float ss = 0.f;
  float4 v[4];
#pragma unroll
  for (int i = 0; i < 4; ++i) {
    v[i] = ((const float4*)p)[lane + 32 * i];
    ss += v[i].x * v[i].x + v[i].y * v[i].y + v[i].z * v[i].z + v[i].w * v[i].w;
  }
#pragma unroll
  for (int off = 1; off < 32; off <<= 1) ss += __shfl_xor(ss, off, 32);
  float sc = 1.0f / fmaxf(sqrtf(ss), 1e-12f);
  _Float16* o = out + (size_t)row * HALF;
#pragma unroll
  for (int i = 0; i < 4; ++i) {
    int e = (lane + 32 * i) * 4;
    o[e + 0] = (_Float16)(v[i].x * sc);
    o[e + 1] = (_Float16)(v[i].y * sc);
    o[e + 2] = (_Float16)(v[i].z * sc);
    o[e + 3] = (_Float16)(v[i].w * sc);
  }
}

// per-token LN stats (shared by content & style extractors). One wave/token.
__global__ __launch_bounds__(256) void k_ln_stats(const float* __restrict__ x,
                                                  float* __restrict__ mu,
                                                  float* __restrict__ rs) {
  int lane = threadIdx.x & 31;
  int tok = blockIdx.x * 8 + (threadIdx.x >> 5);
  const float* p = x + (size_t)tok * DMODEL;
  float s = 0.f, ss = 0.f;
#pragma unroll
  for (int i = 0; i < 8; ++i) {
    float4 v = ((const float4*)p)[lane + 32 * i];
    s += v.x + v.y + v.z + v.w;
    ss += v.x * v.x + v.y * v.y + v.z * v.z + v.w * v.w;
  }
#pragma unroll
  for (int off = 1; off < 32; off <<= 1) {
    s += __shfl_xor(s, off, 32);
    ss += __shfl_xor(ss, off, 32);
  }
  if (lane == 0) {
    float m = s * (1.0f / DMODEL);
    mu[tok] = m;
    rs[tok] = rsqrtf(ss * (1.0f / DMODEL) - m * m + 1e-5f);
  }
}

__global__ void k_zero_acc(float* a) {
  if (threadIdx.x < 4) a[threadIdx.x] = 0.f;
}

// ------- projection GEMM: f16out = tanh(LN(x)*W^T + b), 32x64 wave tile -------
__global__ __launch_bounds__(256) void k_proj_gemm(
    const float* __restrict__ x, const float* __restrict__ mu,
    const float* __restrict__ rs, const float* __restrict__ g,
    const float* __restrict__ bln, const _Float16* __restrict__ W,
    const float* __restrict__ bias, _Float16* __restrict__ out) {
  int lane = threadIdx.x & 31;
  int wave = threadIdx.x >> 5;
  int row0 = blockIdx.x * 256 + wave * 32;   // 8 waves * 32 rows
  int nb0 = blockIdx.y * 64;
  int m = lane & 15, hh = lane >> 4;
  const float* xrow0 = x + (size_t)(row0 + m) * DMODEL;
  const float* xrow1 = x + (size_t)(row0 + 16 + m) * DMODEL;
  float mu0 = mu[row0 + m], rs0 = rs[row0 + m];
  float mu1 = mu[row0 + 16 + m], rs1 = rs[row0 + 16 + m];
  v8f acc[2][4] = {};
  for (int k = 0; k < DMODEL; k += 32) {
    __builtin_prefetch((const void*)(xrow0 + k + 64), 0, 1);
    // gamma/beta for this lane's K-slice, shared by both row-groups
    int kk = k + 8 * hh;
    float4 gv[4], bv[4];
#pragma unroll
    for (int part = 0; part < 2; ++part)
#pragma unroll
      for (int q = 0; q < 2; ++q) {
        gv[part * 2 + q] = ((const float4*)(g + kk + part * 16))[q];
        bv[part * 2 + q] = ((const float4*)(bln + kk + part * 16))[q];
      }
    v16h a0 = make_a_ln(xrow0, mu0, rs0, gv, bv, k, hh);
    v16h a1 = make_a_ln(xrow1, mu1, rs1, gv, bv, k, hh);
#pragma unroll
    for (int t = 0; t < 4; ++t) {
      v16h b = load_b_f16(W, DMODEL, nb0 + 16 * t, k, lane);
      acc[0][t] = wmma32(a0, b, acc[0][t]);
      acc[1][t] = wmma32(a1, b, acc[1][t]);
    }
  }
  int n = lane & 15;
#pragma unroll
  for (int t = 0; t < 4; ++t) {
    int col = nb0 + 16 * t + n;
    float bvv = bias[col];
#pragma unroll
    for (int rg = 0; rg < 2; ++rg)
#pragma unroll
      for (int v = 0; v < 8; ++v) {
        int r = row0 + 16 * rg + v + 8 * hh;
        out[(size_t)r * HALF + col] = (_Float16)tanhf(acc[rg][t][v] + bvv);
      }
  }
}

// ---- VQ: argmax over codes of F·E^T (== argmin cosine dist), 32x64 tile -----
__global__ __launch_bounds__(256) void k_vq_argmax(
    const _Float16* __restrict__ F, const _Float16* __restrict__ E,
    int* __restrict__ idxOut) {
  int lane = threadIdx.x & 31;
  int wave = threadIdx.x >> 5;
  int row0 = blockIdx.x * 256 + wave * 32;
  int n = lane & 15, hh = lane >> 4;
  float best[2][8];
  int bidx[2][8];
#pragma unroll
  for (int rg = 0; rg < 2; ++rg)
#pragma unroll
    for (int v = 0; v < 8; ++v) { best[rg][v] = -1e30f; bidx[rg][v] = 0; }

  for (int c0 = 0; c0 < NCODES; c0 += 64) {
    v8f acc[2][4] = {};
    for (int k = 0; k < HALF; k += 32) {
      v16h a0 = load_a_f16(F, HALF, row0, k, lane);
      v16h a1 = load_a_f16(F, HALF, row0 + 16, k, lane);
      __builtin_prefetch((const void*)(E + (size_t)(c0 + 64) * HALF + k), 0, 1);
#pragma unroll
      for (int t = 0; t < 4; ++t) {
        v16h b = load_b_f16(E, HALF, c0 + 16 * t, k, lane);
        acc[0][t] = wmma32(a0, b, acc[0][t]);
        acc[1][t] = wmma32(a1, b, acc[1][t]);
      }
    }
#pragma unroll
    for (int t = 0; t < 4; ++t) {
      int col = c0 + 16 * t + n;
#pragma unroll
      for (int rg = 0; rg < 2; ++rg)
#pragma unroll
        for (int v = 0; v < 8; ++v) {
          float val = acc[rg][t][v];
          if (val > best[rg][v]) { best[rg][v] = val; bidx[rg][v] = col; }
        }
    }
  }
  // reduce (val,idx) across the 16 lanes of each half-wave; first-index tiebreak
#pragma unroll
  for (int rg = 0; rg < 2; ++rg)
#pragma unroll
    for (int v = 0; v < 8; ++v) {
      float val = best[rg][v];
      int id = bidx[rg][v];
#pragma unroll
      for (int off = 1; off < 16; off <<= 1) {
        float ov = __shfl_xor(val, off, 32);
        int oi = __shfl_xor(id, off, 32);
        if (ov > val || (ov == val && oi < id)) { val = ov; id = oi; }
      }
      if ((lane & 15) == 0) idxOut[row0 + 16 * rg + v + 8 * hh] = id;
    }
}

// ---------------- gather quantized rows + loss partials ----------------
__device__ __forceinline__ float block_sum256(float v, float* sh) {
  sh[threadIdx.x] = v;
  __syncthreads();
  for (int s = 128; s > 0; s >>= 1) {
    if (threadIdx.x < s) sh[threadIdx.x] += sh[threadIdx.x + s];
    __syncthreads();
  }
  float r = sh[0];
  __syncthreads();
  return r;
}

__global__ __launch_bounds__(256) void k_gather_loss(
    const _Float16* __restrict__ cnt, const _Float16* __restrict__ sty,
    const _Float16* __restrict__ cbc, const _Float16* __restrict__ cbs,
    const int* __restrict__ idxc, const int* __restrict__ idxs,
    _Float16* __restrict__ quant, float* __restrict__ accum,
    float* __restrict__ outF) {
  __shared__ float sh[256];
  int i = blockIdx.x;
  int tid = threadIdx.x;
  int ic = idxc[i], is_ = idxs[i];
  float d2c = 0.f, d2s = 0.f, dot = 0.f, nc = 0.f, ns = 0.f;
  int e = tid * 2;
#pragma unroll
  for (int j = 0; j < 2; ++j) {
    _Float16 hc = cbc[(size_t)ic * HALF + e + j];
    _Float16 hs = cbs[(size_t)is_ * HALF + e + j];
    float fc = (float)cnt[(size_t)i * HALF + e + j];
    float fs = (float)sty[(size_t)i * HALF + e + j];
    float qc = (float)hc, qs = (float)hs;
    d2c += (qc - fc) * (qc - fc);
    d2s += (qs - fs) * (qs - fs);
    dot += fc * fs;
    nc += fc * fc;
    ns += fs * fs;
    quant[(size_t)i * DMODEL + e + j] = hc;
    quant[(size_t)i * DMODEL + HALF + e + j] = hs;
  }
  d2c = block_sum256(d2c, sh);
  d2s = block_sum256(d2s, sh);
  dot = block_sum256(dot, sh);
  nc = block_sum256(nc, sh);
  ns = block_sum256(ns, sh);
  if (tid == 0) {
    float cs = fabsf(dot) /
               (fmaxf(sqrtf(nc), 1e-12f) * fmaxf(sqrtf(ns), 1e-12f));
    atomicAdd(&accum[0], d2c);
    atomicAdd(&accum[1], d2s);
    atomicAdd(&accum[2], cs);
    outF[OUT_IDXC + i] = (float)ic;
    outF[OUT_IDXS + i] = (float)is_;
  }
}

// ------- combiner GEMM: out_f32 = quant * Wcomb^T + b, 32x64 wave tile -------
__global__ __launch_bounds__(256) void k_combine_gemm(
    const _Float16* __restrict__ Q, const _Float16* __restrict__ W,
    const float* __restrict__ bias, float* __restrict__ out) {
  int lane = threadIdx.x & 31;
  int wave = threadIdx.x >> 5;
  int row0 = blockIdx.x * 256 + wave * 32;
  int nb0 = blockIdx.y * 64;
  int hh = lane >> 4;
  v8f acc[2][4] = {};
  for (int k = 0; k < DMODEL; k += 32) {
    v16h a0 = load_a_f16(Q, DMODEL, row0, k, lane);
    v16h a1 = load_a_f16(Q, DMODEL, row0 + 16, k, lane);
    __builtin_prefetch((const void*)(W + (size_t)nb0 * DMODEL + k + 64), 0, 1);
#pragma unroll
    for (int t = 0; t < 4; ++t) {
      v16h b = load_b_f16(W, DMODEL, nb0 + 16 * t, k, lane);
      acc[0][t] = wmma32(a0, b, acc[0][t]);
      acc[1][t] = wmma32(a1, b, acc[1][t]);
    }
  }
  int n = lane & 15;
#pragma unroll
  for (int t = 0; t < 4; ++t) {
    int col = nb0 + 16 * t + n;
    float bv = bias[col];
#pragma unroll
    for (int rg = 0; rg < 2; ++rg)
#pragma unroll
      for (int v = 0; v < 8; ++v) {
        int r = row0 + 16 * rg + v + 8 * hh;
        out[(size_t)r * DMODEL + col] = acc[rg][t][v] + bv;
      }
  }
}

// ---------------- output LayerNorm (in place on d_out). One wave/token -------
__global__ __launch_bounds__(256) void k_out_ln(float* __restrict__ out,
                                                const float* __restrict__ g,
                                                const float* __restrict__ b) {
  int lane = threadIdx.x & 31;
  int tok = blockIdx.x * 8 + (threadIdx.x >> 5);
  float* p = out + (size_t)tok * DMODEL;
  float4 v[8];
  float s = 0.f, ss = 0.f;
#pragma unroll
  for (int i = 0; i < 8; ++i) {
    v[i] = ((const float4*)p)[lane + 32 * i];
    s += v[i].x + v[i].y + v[i].z + v[i].w;
    ss += v[i].x * v[i].x + v[i].y * v[i].y + v[i].z * v[i].z + v[i].w * v[i].w;
  }
#pragma unroll
  for (int off = 1; off < 32; off <<= 1) {
    s += __shfl_xor(s, off, 32);
    ss += __shfl_xor(ss, off, 32);
  }
  float m = s * (1.0f / DMODEL);
  float r = rsqrtf(ss * (1.0f / DMODEL) - m * m + 1e-5f);
#pragma unroll
  for (int i = 0; i < 8; ++i) {
    int e = (lane + 32 * i) * 4;
    float4 gv = ((const float4*)(g + e))[0];
    float4 bv = ((const float4*)(b + e))[0];
    float4 o;
    o.x = (v[i].x - m) * r * gv.x + bv.x;
    o.y = (v[i].y - m) * r * gv.y + bv.y;
    o.z = (v[i].z - m) * r * gv.z + bv.z;
    o.w = (v[i].w - m) * r * gv.w + bv.w;
    ((float4*)p)[lane + 32 * i] = o;
  }
}

__global__ void k_finalize(const float* __restrict__ accum,
                           float* __restrict__ out) {
  const float invNH = 1.0f / ((float)NTOK * (float)HALF);
  float dis = accum[2] * (1.0f / (float)NTOK);
  dis = fminf(fmaxf(dis, 0.0f), 1.0f);
  float total = 0.1f * accum[0] * invNH + 0.1f * accum[1] * invNH + 0.5f * dis;
  out[OUT_TOTAL] = total;
  out[OUT_DIS] = dis;
}

// ---------------- host launcher ----------------
extern "C" void kernel_launch(void* const* d_in, const int* in_sizes, int n_in,
                              void* d_out, int out_size, void* d_ws,
                              size_t ws_size, hipStream_t stream) {
  (void)in_sizes; (void)n_in; (void)out_size; (void)ws_size;
  const float* x      = (const float*)d_in[0];
  const float* ln_c_g = (const float*)d_in[1];
  const float* ln_c_b = (const float*)d_in[2];
  const float* W_c    = (const float*)d_in[3];
  const float* b_c    = (const float*)d_in[4];
  const float* ln_s_g = (const float*)d_in[5];
  const float* ln_s_b = (const float*)d_in[6];
  const float* W_s    = (const float*)d_in[7];
  const float* b_s    = (const float*)d_in[8];
  const float* cb_c   = (const float*)d_in[9];
  const float* cb_s   = (const float*)d_in[10];
  const float* W_comb = (const float*)d_in[11];
  const float* b_comb = (const float*)d_in[12];
  const float* ln_o_g = (const float*)d_in[13];
  const float* ln_o_b = (const float*)d_in[14];

  char* ws = (char*)d_ws;
  _Float16* wc_h    = (_Float16*)(ws + OFF_WC);
  _Float16* ws_h    = (_Float16*)(ws + OFF_WS);
  _Float16* wcomb_h = (_Float16*)(ws + OFF_WCOMB);
  _Float16* cbc_h   = (_Float16*)(ws + OFF_CBC);
  _Float16* cbs_h   = (_Float16*)(ws + OFF_CBS);
  _Float16* cnt_h   = (_Float16*)(ws + OFF_CNT);
  _Float16* sty_h   = (_Float16*)(ws + OFF_STY);
  _Float16* quant_h = (_Float16*)(ws + OFF_QUANT);
  float*    mu      = (float*)(ws + OFF_MU);
  float*    rs      = (float*)(ws + OFF_RS);
  int*      idxc    = (int*)(ws + OFF_IDXC);
  int*      idxs    = (int*)(ws + OFF_IDXS);
  float*    accum   = (float*)(ws + OFF_ACC);
  float*    outF    = (float*)d_out;

  // 1) weight conversion to f16
  k_convert_f16<<<(HALF * DMODEL + 255) / 256, 256, 0, stream>>>(W_c, wc_h, HALF * DMODEL);
  k_convert_f16<<<(HALF * DMODEL + 255) / 256, 256, 0, stream>>>(W_s, ws_h, HALF * DMODEL);
  k_convert_f16<<<(DMODEL * DMODEL + 255) / 256, 256, 0, stream>>>(W_comb, wcomb_h, DMODEL * DMODEL);
  // 2) codebook L2-normalize -> f16
  k_norm_cb<<<NCODES / 8, 256, 0, stream>>>(cb_c, cbc_h);
  k_norm_cb<<<NCODES / 8, 256, 0, stream>>>(cb_s, cbs_h);
  // 3) shared LN stats, zero loss accumulators
  k_ln_stats<<<NTOK / 8, 256, 0, stream>>>(x, mu, rs);
  k_zero_acc<<<1, 32, 0, stream>>>(accum);
  // 4) fused LN + projection GEMMs + tanh (WMMA, 32x64 wave tiles)
  dim3 gProj(NTOK / 256, HALF / 64);
  k_proj_gemm<<<gProj, 256, 0, stream>>>(x, mu, rs, ln_c_g, ln_c_b, wc_h, b_c, cnt_h);
  k_proj_gemm<<<gProj, 256, 0, stream>>>(x, mu, rs, ln_s_g, ln_s_b, ws_h, b_s, sty_h);
  // 5) VQ nearest-code search (WMMA GEMM + fused argmax, 32x64 wave tiles)
  k_vq_argmax<<<NTOK / 256, 256, 0, stream>>>(cnt_h, cbc_h, idxc);
  k_vq_argmax<<<NTOK / 256, 256, 0, stream>>>(sty_h, cbs_h, idxs);
  // 6) gather quantized rows, loss partials, index outputs
  k_gather_loss<<<NTOK, 256, 0, stream>>>(cnt_h, sty_h, cbc_h, cbs_h, idxc, idxs,
                                          quant_h, accum, outF);
  // 7) combiner GEMM (WMMA) -> pre-LN fp32 in d_out
  dim3 gComb(NTOK / 256, DMODEL / 64);
  k_combine_gemm<<<gComb, 256, 0, stream>>>(quant_h, wcomb_h, b_comb, outF);
  // 8) output LayerNorm in place
  k_out_ln<<<NTOK / 8, 256, 0, stream>>>(outF, ln_o_g, ln_o_b);
  // 9) scalar losses
  k_finalize<<<1, 1, 0, stream>>>(accum, outF);
}
